// RSSA_3848290697745
// MI455X (gfx1250) — matrix operand order repaired and checked by reading, not compiled, not run on hardware
//
#include <hip/hip_runtime.h>

#define T_  4
#define B_  32
#define C_  512
#define N_  256
#define NH_ 8
#define HD_ 64

typedef __attribute__((ext_vector_type(16))) _Float16 v16h;
typedef __attribute__((ext_vector_type(8)))  _Float16 v8h;
typedef __attribute__((ext_vector_type(8)))  float    v8f;
typedef __attribute__((ext_vector_type(4)))  int      v4i;

#define AS1 __attribute__((address_space(1)))
#define AS3 __attribute__((address_space(3)))

#if __has_builtin(__builtin_amdgcn_global_load_async_to_lds_b128) && \
    __has_builtin(__builtin_amdgcn_s_wait_asynccnt)
#define HAVE_ASYNC_LDS 1
#else
#define HAVE_ASYNC_LDS 0
#endif

static __device__ __forceinline__ v16h cat16(v8h a, v8h b) {
  return __builtin_shufflevector(a, b, 0,1,2,3,4,5,6,7,8,9,10,11,12,13,14,15);
}

// A fragment (16x32 f16). Element (m,k) at base[(row0+m)*stride + k].
// Lane L (0..15): row=row0+L, halves 0..7 -> K=kb..kb+7, halves 8..15 -> K=kb+16..kb+23.
// Lane 16+L: same row set, K offset +8 (per CDNA5 ISA 16-bit A layout).
static __device__ __forceinline__ v16h load_a_frag(const _Float16* base, int stride,
                                                   int row0, int kb, int lane) {
  const _Float16* p = base + (size_t)(row0 + (lane & 15)) * stride + kb + ((lane >> 4) << 3);
  v8h lo = *(const v8h*)(p);
  v8h hi = *(const v8h*)(p + 16);
  return cat16(lo, hi);
}

// B fragment (32x16 f16). Element (k,n) at base[(col0+n)*stride + k] (K contiguous per column).
// Lane L (0..15): col=col0+L, halves 0..7 -> K=kb..kb+7, halves 8..15 -> K=kb+16..kb+23; lane 16+L: K offset +8.
static __device__ __forceinline__ v16h load_b_frag(const _Float16* base, int stride,
                                                   int col0, int kb, int lane) {
  const _Float16* p = base + (size_t)(col0 + (lane & 15)) * stride + kb + ((lane >> 4) << 3);
  v8h lo = *(const v8h*)(p);
  v8h hi = *(const v8h*)(p + 16);
  return cat16(lo, hi);
}

static __device__ __forceinline__ v8f wmma_f16(v16h a, v16h b, v8f c) {
  return __builtin_amdgcn_wmma_f32_16x16x32_f16(false, a, false, b, (short)0, c, false, false);
}

// ---------------- prep: f32 weights -> f16 (wq,wk,wv,wp contiguous) ----------------
__global__ void rssa_prep_w(const float* __restrict__ wq, const float* __restrict__ wk,
                            const float* __restrict__ wv, const float* __restrict__ wp,
                            _Float16* __restrict__ w16) {
  int idx = blockIdx.x * blockDim.x + threadIdx.x;       // 4*C*C = 1048576
  int which = idx >> 18;                                  // C*C = 262144 = 2^18
  int i = idx & (C_ * C_ - 1);
  const float* src = (which == 0) ? wq : (which == 1) ? wk : (which == 2) ? wv : wp;
  w16[idx] = (_Float16)src[i];
}

// ---------------- prep: fold BN (and bias for P branch) into inv/off per channel ----------------
__global__ void rssa_prep_bn(const float* __restrict__ bnq, const float* __restrict__ bnk,
                             const float* __restrict__ bnv, const float* __restrict__ bnp,
                             const float* __restrict__ bp, float* __restrict__ bnc) {
  int c = blockIdx.x * blockDim.x + threadIdx.x;
  if (c >= C_) return;
  const float* bns[4] = {bnq, bnk, bnv, bnp};
#pragma unroll
  for (int j = 0; j < 4; ++j) {
    float g  = bns[j][c];
    float be = bns[j][C_ + c];
    float mn = bns[j][2 * C_ + c];
    float va = bns[j][3 * C_ + c];
    float inv = g * rsqrtf(va + 1e-5f);
    float off = be - mn * inv;
    if (j == 3) off = bp[c] * inv + off;   // fold conv bias through BN
    bnc[(2 * j) * C_ + c]     = inv;
    bnc[(2 * j + 1) * C_ + c] = off;
  }
}

// ---------------- input LIF: x[t,b,c,n] -> spikes xsT[t,b,n,c] (f16, c contiguous) ----------------
__global__ void rssa_lif_in(const float* __restrict__ x, _Float16* __restrict__ xsT) {
  int tid = blockIdx.x * blockDim.x + threadIdx.x;       // B*C*N = 4194304
  int n  = tid & (N_ - 1);
  int bc = tid >> 8;
  int b  = bc >> 9;
  int c  = bc & (C_ - 1);
  float v = 0.0f;
#pragma unroll
  for (int t = 0; t < T_; ++t) {
    float xv = x[((size_t)(t * B_ + b) * C_ + c) * N_ + n];
    v = 0.5f * v + xv;
    float s = (v >= 1.0f) ? 1.0f : 0.0f;
    v *= (1.0f - s);
    xsT[((size_t)(t * B_ + b) * N_ + n) * C_ + c] = (_Float16)s;
  }
}

// ---------------- fused QKV GEMM + BN + LIF -> spike tensors ----------------
// grid: (B*N/16, C/16/8, 3 branches), block 256 (8 waves); each wave: one 16(d)x16(n) tile, all 4 t.
// q_s: [t,b,h,n,hd] (hd contig)   k_s,v_s: [t,b,h,hd,n] (n contig)
__global__ __launch_bounds__(256) void rssa_qkv(const _Float16* __restrict__ w16,
                                                const _Float16* __restrict__ xsT,
                                                const float* __restrict__ bnc,
                                                _Float16* __restrict__ q_s,
                                                _Float16* __restrict__ k_s,
                                                _Float16* __restrict__ v_s) {
  int br   = blockIdx.z;
  const _Float16* W = w16 + (size_t)br * C_ * C_;
  const float* inv = bnc + (size_t)(2 * br) * C_;
  const float* off = bnc + (size_t)(2 * br + 1) * C_;
  _Float16* dst = (br == 0) ? q_s : (br == 1) ? k_s : v_s;

  int wave = threadIdx.x >> 5, lane = threadIdx.x & 31;
  int d0  = (blockIdx.y * 8 + wave) * 16;
  int b   = blockIdx.x >> 4;
  int n0  = (blockIdx.x & 15) << 4;
  int col = n0 + (lane & 15);

  float invr[8], offr[8];
#pragma unroll
  for (int r = 0; r < 8; ++r) {
    int d = d0 + r + ((lane >> 4) << 3);
    invr[r] = inv[d];
    offr[r] = off[d];
  }

  v8f acc[T_] = {};
  for (int ks = 0; ks < C_ / 32; ++ks) {
    int kb = ks * 32;
    // prefetch next k-step fragments (global_prefetch_b8)
    if (ks + 1 < C_ / 32) {
      __builtin_prefetch(W + (size_t)(d0 + (lane & 15)) * C_ + kb + 32, 0, 1);
      __builtin_prefetch(xsT + (size_t)b * N_ * C_ + (size_t)col * C_ + kb + 32, 0, 1);
    }
    v16h a = load_a_frag(W, C_, d0, kb, lane);
#pragma unroll
    for (int t = 0; t < T_; ++t) {
      const _Float16* bsrc = xsT + (size_t)(t * B_ + b) * N_ * C_;
      v16h bb = load_b_frag(bsrc, C_, col, kb, lane);
      acc[t] = wmma_f16(a, bb, acc[t]);
    }
  }

  float vst[8] = {};
#pragma unroll
  for (int t = 0; t < T_; ++t) {
#pragma unroll
    for (int r = 0; r < 8; ++r) {
      int d = d0 + r + ((lane >> 4) << 3);
      float y = acc[t][r] * invr[r] + offr[r];
      float v = 0.5f * vst[r] + y;
      float s = (v >= 1.0f) ? 1.0f : 0.0f;
      vst[r] = v * (1.0f - s);
      int hh = d >> 6, i = d & 63;
      size_t o;
      if (br == 0) o = (((size_t)(t * B_ + b) * NH_ + hh) * N_ + col) * HD_ + i;
      else         o = (((size_t)(t * B_ + b) * NH_ + hh) * HD_ + i) * N_ + col;
      dst[o] = (_Float16)s;
    }
  }
}

// ---------------- linear attention per (b,h): G = m_k^T v_prev + k_t^T v_t ; out = q G * scale ----
// grid 256 (= B*NH), block 256. LDS: EMA key memory (f16) + scaled G^T.
__global__ __launch_bounds__(256) void rssa_attn(const _Float16* __restrict__ q_s,
                                                 const _Float16* __restrict__ k_s,
                                                 const _Float16* __restrict__ v_s,
                                                 const float* __restrict__ betap,
                                                 _Float16* __restrict__ a_s) {
  __shared__ __align__(16) _Float16 mk[HD_ * N_];     // 32 KB: EMA key memory [hd][n]
  __shared__ __align__(16) _Float16 Gt[HD_ * 72];     // scaled G^T [j][i], padded stride 72

  int bh = blockIdx.x;
  int b = bh >> 3, h = bh & 7;
  int wave = threadIdx.x >> 5, lane = threadIdx.x & 31;
  float beta = betap[0];

  const size_t hsz = (size_t)HD_ * N_;
  float st[8][8] = {};   // attn-LIF membrane state per output tile element

  for (int t = 0; t < T_; ++t) {
    size_t ho  = ((size_t)(t * B_ + b) * NH_ + h) * hsz;
    size_t hop = (t > 0) ? (((size_t)((t - 1) * B_ + b) * NH_ + h) * hsz) : ho;

    // ---- Gram phase: 16 tiles of 64x64 G; 2 per wave ----
#pragma unroll
    for (int q2 = 0; q2 < 2; ++q2) {
      int idx = wave * 2 + q2;
      int ti = idx >> 2, tj = idx & 3;
      v8f g = {};
      if (t == 0) {
        const _Float16* A = k_s + ho;
        const _Float16* Bv = v_s + ho;
#pragma unroll
        for (int ks = 0; ks < N_ / 32; ++ks) {
          v16h a  = load_a_frag(A,  N_, ti * 16, ks * 32, lane);
          v16h bb = load_b_frag(Bv, N_, tj * 16, ks * 32, lane);
          g = wmma_f16(a, bb, g);
        }
      } else {
        const _Float16* Bp = v_s + hop;      // v_prev
#pragma unroll
        for (int ks = 0; ks < N_ / 32; ++ks) {
          v16h a  = load_a_frag(mk, N_, ti * 16, ks * 32, lane);   // m_k (LDS)
          v16h bb = load_b_frag(Bp, N_, tj * 16, ks * 32, lane);
          g = wmma_f16(a, bb, g);
        }
        const _Float16* A2 = k_s + ho;
        const _Float16* B2 = v_s + ho;
#pragma unroll
        for (int ks = 0; ks < N_ / 32; ++ks) {
          v16h a  = load_a_frag(A2, N_, ti * 16, ks * 32, lane);
          v16h bb = load_b_frag(B2, N_, tj * 16, ks * 32, lane);
          g = wmma_f16(a, bb, g);
        }
      }
      // store scaled G transposed: Gt[j][i] = 0.125 * G[i][j]
#pragma unroll
      for (int r = 0; r < 8; ++r) {
        int i = ti * 16 + r + ((lane >> 4) << 3);
        int j = tj * 16 + (lane & 15);
        Gt[j * 72 + i] = (_Float16)(g[r] * 0.125f);
      }
    }
    __syncthreads();

    // ---- EMA key-memory maintenance (runs alongside out phase; disjoint LDS) ----
    if (t == 0) {
#if HAVE_ASYNC_LDS
      // async global -> LDS copy of k0 into EMA memory (ASYNCcnt path)
      AS1 v4i* src = (AS1 v4i*)(void*)(k_s + ho);
      AS3 v4i* dl  = (AS3 v4i*)(void*)mk;
#pragma unroll
      for (int e = 0; e < 8; ++e) {
        int idx16 = e * 256 + (int)threadIdx.x;
        __builtin_amdgcn_global_load_async_to_lds_b128(src + idx16, dl + idx16, 0, 0);
      }
      __builtin_amdgcn_s_wait_asynccnt(0);
#else
      const uint4* src = (const uint4*)(k_s + ho);
      uint4* dmk = (uint4*)mk;
#pragma unroll
      for (int e = 0; e < 8; ++e) dmk[e * 256 + threadIdx.x] = src[e * 256 + threadIdx.x];
#endif
    } else {
      const _Float16* kt = k_s + ho;
#pragma unroll 4
      for (int e = 0; e < 64; ++e) {
        int ii = e * 256 + threadIdx.x;
        float m = (float)mk[ii];
        mk[ii] = (_Float16)(beta * m + (1.0f - beta) * (float)kt[ii]);
      }
    }

    // ---- out phase: out[256,64] = q_t @ G_scaled ; fused attn-LIF (thr=0.5) ----
    const _Float16* Q = q_s + ho;   // [n][hd]
#pragma unroll
    for (int rt2 = 0; rt2 < 2; ++rt2) {
      int rt = wave * 2 + rt2;
#pragma unroll
      for (int ct = 0; ct < 4; ++ct) {
        v8f o = {};
#pragma unroll
        for (int ks = 0; ks < 2; ++ks) {
          v16h a  = load_a_frag(Q,  HD_, rt * 16, ks * 32, lane);
          v16h bb = load_b_frag(Gt, 72,  ct * 16, ks * 32, lane);
          o = wmma_f16(a, bb, o);
        }
        int si = rt2 * 4 + ct;
#pragma unroll
        for (int r = 0; r < 8; ++r) {
          int n = rt * 16 + r + ((lane >> 4) << 3);
          int j = ct * 16 + (lane & 15);
          float v = 0.5f * st[si][r] + o[r];
          float s = (v >= 0.5f) ? 1.0f : 0.0f;
          st[si][r] = v * (1.0f - s);
          a_s[((size_t)(t * B_ + b) * N_ + n) * C_ + (h * HD_ + j)] = (_Float16)s;
        }
      }
    }
    __syncthreads();
  }
}

// ---------------- final projection: out = BN(wp @ a + bp), f32 output [t,b,c,n] ----------------
__global__ __launch_bounds__(256) void rssa_proj(const _Float16* __restrict__ wp16,
                                                 const _Float16* __restrict__ a_s,
                                                 const float* __restrict__ bnc,
                                                 float* __restrict__ out) {
  const float* inv = bnc + 6 * C_;
  const float* off = bnc + 7 * C_;   // includes folded bias
  int wave = threadIdx.x >> 5, lane = threadIdx.x & 31;
  int d0  = (blockIdx.y * 8 + wave) * 16;
  int b   = blockIdx.x >> 4;
  int n0  = (blockIdx.x & 15) << 4;
  int col = n0 + (lane & 15);

  float invr[8], offr[8];
#pragma unroll
  for (int r = 0; r < 8; ++r) {
    int d = d0 + r + ((lane >> 4) << 3);
    invr[r] = inv[d];
    offr[r] = off[d];
  }

  v8f acc[T_] = {};
  for (int ks = 0; ks < C_ / 32; ++ks) {
    int kb = ks * 32;
    if (ks + 1 < C_ / 32) {
      __builtin_prefetch(wp16 + (size_t)(d0 + (lane & 15)) * C_ + kb + 32, 0, 1);
      __builtin_prefetch(a_s + (size_t)b * N_ * C_ + (size_t)col * C_ + kb + 32, 0, 1);
    }
    v16h a = load_a_frag(wp16, C_, d0, kb, lane);
#pragma unroll
    for (int t = 0; t < T_; ++t) {
      const _Float16* bsrc = a_s + (size_t)(t * B_ + b) * N_ * C_;
      v16h bb = load_b_frag(bsrc, C_, col, kb, lane);
      acc[t] = wmma_f16(a, bb, acc[t]);
    }
  }
#pragma unroll
  for (int t = 0; t < T_; ++t) {
#pragma unroll
    for (int r = 0; r < 8; ++r) {
      int d = d0 + r + ((lane >> 4) << 3);
      out[((size_t)(t * B_ + b) * C_ + d) * N_ + col] = acc[t][r] * invr[r] + offr[r];
    }
  }
}

// ---------------- workspace layout (bytes) ----------------
static const size_t XS_OFF  = 0;                         // [T,B,N,C] f16     33,554,432
static const size_t W16_OFF = XS_OFF  + 33554432;        // 4x[C,C] f16        2,097,152
static const size_t Q_OFF   = W16_OFF + 2097152;         // [T,B,h,N,hd] f16  33,554,432
static const size_t K_OFF   = Q_OFF   + 33554432;        // [T,B,h,hd,N] f16
static const size_t V_OFF   = K_OFF   + 33554432;
static const size_t A_OFF   = V_OFF   + 33554432;        // [T,B,N,C] f16
static const size_t BNC_OFF = A_OFF   + 33554432;        // 8x[C] f32             16,384
static const size_t WS_NEED = BNC_OFF + 16384;           // ~162 MB total

extern "C" void kernel_launch(void* const* d_in, const int* in_sizes, int n_in,
                              void* d_out, int out_size, void* d_ws, size_t ws_size,
                              hipStream_t stream) {
  (void)in_sizes; (void)n_in; (void)out_size;
  if (ws_size < WS_NEED) return;

  const float* x    = (const float*)d_in[0];
  const float* wq   = (const float*)d_in[1];
  const float* wk   = (const float*)d_in[2];
  const float* wv   = (const float*)d_in[3];
  const float* wp   = (const float*)d_in[4];
  const float* bp   = (const float*)d_in[5];
  const float* beta = (const float*)d_in[6];
  const float* bnq  = (const float*)d_in[7];
  const float* bnk  = (const float*)d_in[8];
  const float* bnv  = (const float*)d_in[9];
  const float* bnp  = (const float*)d_in[10];

  char* ws = (char*)d_ws;
  _Float16* xsT = (_Float16*)(ws + XS_OFF);
  _Float16* w16 = (_Float16*)(ws + W16_OFF);
  _Float16* q_s = (_Float16*)(ws + Q_OFF);
  _Float16* k_s = (_Float16*)(ws + K_OFF);
  _Float16* v_s = (_Float16*)(ws + V_OFF);
  _Float16* a_s = (_Float16*)(ws + A_OFF);
  float*    bnc = (float*)(ws + BNC_OFF);
  float*    out = (float*)d_out;

  rssa_prep_w <<<4096, 256, 0, stream>>>(wq, wk, wv, wp, w16);
  rssa_prep_bn<<<2,    256, 0, stream>>>(bnq, bnk, bnv, bnp, bp, bnc);
  rssa_lif_in <<<16384, 256, 0, stream>>>(x, xsT);
  rssa_qkv    <<<dim3(512, 4, 3), 256, 0, stream>>>(w16, xsT, bnc, q_s, k_s, v_s);
  rssa_attn   <<<256, 256, 0, stream>>>(q_s, k_s, v_s, beta, a_s);
  rssa_proj   <<<dim3(512, 4, 1), 256, 0, stream>>>(w16 + 3 * C_ * C_, a_s, bnc, out);
}
